// ChebConv_38809324486714
// MI455X (gfx1250) — matrix operand order repaired and checked by reading, not compiled, not run on hardware
//
#include <hip/hip_runtime.h>
#include <hip/hip_bf16.h>

typedef __attribute__((ext_vector_type(16))) __bf16 v16bf;
typedef __attribute__((ext_vector_type(8)))  float  v8f;

#define C_DIM 128

// ---------------------------------------------------------------------------
// bf16 split helpers (native hardware converts, RNE)
// ---------------------------------------------------------------------------
__device__ __forceinline__ void bf16_split(float x, __bf16& hi, __bf16& lo) {
    hi = (__bf16)x;
    lo = (__bf16)(x - (float)hi);
}

// ---------------------------------------------------------------------------
// Kernel 1: zero the aggregation workspace ([2N,128] f32) with float4 stores
// ---------------------------------------------------------------------------
__global__ void zero_agg_kernel(float* __restrict__ p, long n4) {
    long i = (long)blockIdx.x * blockDim.x + threadIdx.x;
    long stride = (long)gridDim.x * blockDim.x;
    float4* p4 = (float4*)p;
    for (; i < n4; i += stride) p4[i] = make_float4(0.f, 0.f, 0.f, 0.f);
}

// ---------------------------------------------------------------------------
// Kernel 2: pre-convert W[128,128] f32 into bf16 hi/lo fragments stored in
// WMMA B-fragment order. Flat index t = ((wt*4+ks)*32 + lane)*16 + i maps to
// W element K = ks*32 + (lane>>4)*8 + (i&7) + (i>>3)*16, n = wt*16 + (lane&15).
// The GEMM hot loop then loads each fragment with two b128 loads, no cvt.
// ---------------------------------------------------------------------------
__global__ void weight_frag_kernel(const float* __restrict__ W,
                                   unsigned short* __restrict__ Whi,
                                   unsigned short* __restrict__ Wlo) {
    int t = blockIdx.x * blockDim.x + threadIdx.x;   // 0 .. 16383
    if (t >= C_DIM * C_DIM) return;
    const int i    = t & 15;
    const int lane = (t >> 4) & 31;
    const int ks   = (t >> 9) & 3;
    const int wt   = t >> 11;
    const int K = ks * 32 + (lane >> 4) * 8 + (i & 7) + (i >> 3) * 16;
    const int n = wt * 16 + (lane & 15);
    const float w = W[(size_t)K * C_DIM + n];
    __bf16 hi, lo;
    bf16_split(w, hi, lo);
    Whi[t] = __builtin_bit_cast(unsigned short, hi);
    Wlo[t] = __builtin_bit_cast(unsigned short, lo);
}

// ---------------------------------------------------------------------------
// Kernel 3: fused complex SpMM over edges (COO gather + atomic scatter-add).
//   agg_real[row] += Lr*Xr[col] - Li*Xi[col]
//   agg_imag[row] += Li*Xr[col] + Lr*Xi[col]
// One wave per edge per iteration (edge index is wave-uniform -> scalar
// loads of COO data); lane l covers channels 4l..4l+3 with float4 gathers.
// X (51 MB) and agg (51 MB) are L2-resident (192 MB L2) -> L2-bound phase.
// ---------------------------------------------------------------------------
__global__ void spmm_complex_kernel(const float* __restrict__ Xr,
                                    const float* __restrict__ Xi,
                                    const float* __restrict__ Lr,
                                    const float* __restrict__ Li,
                                    const int*   __restrict__ row,
                                    const int*   __restrict__ col,
                                    float* __restrict__ aggr,
                                    float* __restrict__ aggi,
                                    int nEdges) {
    const int lane = threadIdx.x & 31;
    const int wave = threadIdx.x >> 5;
    const int wavesPerBlock = blockDim.x >> 5;
    const int waveStride = gridDim.x * wavesPerBlock;

    for (int e = blockIdx.x * wavesPerBlock + wave; e < nEdges; e += waveStride) {
        const int   c  = col[e];
        const int   r  = row[e];
        const float lr = Lr[e];
        const float li = Li[e];

        const float4* xr4 = (const float4*)(Xr + (size_t)c * C_DIM);
        const float4* xi4 = (const float4*)(Xi + (size_t)c * C_DIM);
        const float4 xr = xr4[lane];
        const float4 xi = xi4[lane];

        const size_t base = (size_t)r * C_DIM + (size_t)lane * 4;
        atomicAdd(aggr + base + 0, lr * xr.x - li * xi.x);
        atomicAdd(aggr + base + 1, lr * xr.y - li * xi.y);
        atomicAdd(aggr + base + 2, lr * xr.z - li * xi.z);
        atomicAdd(aggr + base + 3, lr * xr.w - li * xi.w);
        atomicAdd(aggi + base + 0, li * xr.x + lr * xi.x);
        atomicAdd(aggi + base + 1, li * xr.y + lr * xi.y);
        atomicAdd(aggi + base + 2, li * xr.z + lr * xi.z);
        atomicAdd(aggi + base + 3, li * xr.w + lr * xi.w);
    }
}

// ---------------------------------------------------------------------------
// Kernel 4: dense projection + residual via WMMA (bf16x3 ~ f32 precision).
//   out[2N,128] = agg[2N,128] @ W[128,128] + concat(X_real, X_imag)
// Block = 256 threads (8 waves). blockIdx.x = 16-row tile; wave w = 16-col
// tile. K=128 -> 4 kstep iterations (unroll 1 to avoid spills); each kstep
// does 3 x v_wmma_f32_16x16x32_bf16: Ahi*Bhi + Ahi*Blo + Alo*Bhi.
//
// Wave32 fragment layouts per CDNA5 ISA 7.12.2:
//   A 16x32 bf16: lane=half*16+m holds row m; elem i -> K = half*8+i (i<8),
//                 16+half*8+(i-8) (i>=8)  [two contiguous 8-float runs]
//   B 32x16 bf16: pre-packed fragments (two b128 loads per kstep per matrix)
//   C/D 16x16 f32: elem r -> row r+8*(lane/16), col lane%16
//
// Residual select is BLOCK-UNIFORM (50000 % 16 == 0: a tile never straddles
// the real/imag boundary) -> hoisted to one uniform base pointer; epilogue
// is 8 plain load/add/store triples, no divergent exec-mask traffic.
// ---------------------------------------------------------------------------
__global__ void proj_wmma_bf16x3_kernel(const float* __restrict__ agg,  // [2N,128]
                                        const unsigned short* __restrict__ Whi,
                                        const unsigned short* __restrict__ Wlo,
                                        const float* __restrict__ Xr,   // [N,128]
                                        const float* __restrict__ Xi,   // [N,128]
                                        float* __restrict__ out,        // [2N,128]
                                        int nNodes) {
    const int lane  = threadIdx.x & 31;
    const int wTile = threadIdx.x >> 5;   // 0..7 : 16-wide column tile
    const int half  = lane >> 4;          // 0/1 : lane half
    const int m     = lane & 15;

    const long rowBase = (long)blockIdx.x * 16;
    const float* aRow = agg + (rowBase + m) * C_DIM;

    v8f acc = {0.f, 0.f, 0.f, 0.f, 0.f, 0.f, 0.f, 0.f};

    #pragma unroll 1
    for (int ks = 0; ks < 4; ++ks) {
        // ---- A fragment: 4 x float4 loads (two contiguous 8-float runs) ----
        const int ka = ks * 32 + half * 8;
        const float4* ap = (const float4*)(aRow + ka);
        const float4 a0 = ap[0];
        const float4 a1 = ap[1];
        const float4* aq = (const float4*)(aRow + ka + 16);
        const float4 a2 = aq[0];
        const float4 a3 = aq[1];
        float a[16] = {a0.x, a0.y, a0.z, a0.w, a1.x, a1.y, a1.z, a1.w,
                       a2.x, a2.y, a2.z, a2.w, a3.x, a3.y, a3.z, a3.w};

        v16bf ahi, alo;
        #pragma unroll
        for (int i = 0; i < 16; ++i) {
            __bf16 h, l;
            bf16_split(a[i], h, l);
            ahi[i] = h;
            alo[i] = l;
        }

        // ---- B fragments: pre-packed, two b128 loads each ----
        const size_t fbase = (size_t)((wTile * 4 + ks) * 32 + lane) * 16;
        const v16bf bhi = *(const v16bf*)(Whi + fbase);
        const v16bf blo = *(const v16bf*)(Wlo + fbase);

        // D = A*B + C : hi*hi + hi*lo + lo*hi  (lo*lo below f32 noise floor)
        acc = __builtin_amdgcn_wmma_f32_16x16x32_bf16(false, ahi, false, bhi,
                                                      (short)0, acc, false, false);
        acc = __builtin_amdgcn_wmma_f32_16x16x32_bf16(false, ahi, false, blo,
                                                      (short)0, acc, false, false);
        acc = __builtin_amdgcn_wmma_f32_16x16x32_bf16(false, alo, false, bhi,
                                                      (short)0, acc, false, false);
    }

    // ---- epilogue: residual add + store (f32), uniform base pointers ----
    // Tile is entirely in the real half (rowBase < nNodes) or entirely in
    // the imag half; pick the residual base once (uniform, no divergence).
    const float* resBase = (rowBase < nNodes)
                               ? (Xr + rowBase * C_DIM)
                               : (Xi + (rowBase - (long)nNodes) * C_DIM);
    float* outBase = out + rowBase * C_DIM;

    const int colOff = wTile * 16 + m;            // 0..127
    #pragma unroll
    for (int r = 0; r < 8; ++r) {
        const int off = (r + 8 * half) * C_DIM + colOff;   // < 16*128
        outBase[off] = acc[r] + resBase[off];
    }
}

// ---------------------------------------------------------------------------
// Launch: zero agg | pack W -> fused complex SpMM -> WMMA projection
// ---------------------------------------------------------------------------
extern "C" void kernel_launch(void* const* d_in, const int* in_sizes, int n_in,
                              void* d_out, int out_size, void* d_ws, size_t ws_size,
                              hipStream_t stream) {
    const float* Xr = (const float*)d_in[0];
    const float* Xi = (const float*)d_in[1];
    const float* Lr = (const float*)d_in[2];
    const float* Li = (const float*)d_in[3];
    const float* W  = (const float*)d_in[4];
    const int* row  = (const int*)d_in[5];
    const int* col  = (const int*)d_in[6];

    const int nNodes = in_sizes[0] / C_DIM;
    const int nEdges = in_sizes[2];

    // workspace layout: agg [2N,128] f32, then Whi/Wlo fragment buffers
    float* agg  = (float*)d_ws;
    float* aggr = agg;
    float* aggi = agg + (size_t)nNodes * C_DIM;
    unsigned short* Whi = (unsigned short*)(agg + (size_t)2 * nNodes * C_DIM);
    unsigned short* Wlo = Whi + (size_t)C_DIM * C_DIM;
    float* out = (float*)d_out;

    // 1) zero aggregation buffers
    const long n4 = (long)2 * nNodes * C_DIM / 4;
    zero_agg_kernel<<<1024, 256, 0, stream>>>(agg, n4);

    // 2) pack weight into bf16 hi/lo B-fragments (64 blocks, independent of 1)
    weight_frag_kernel<<<(C_DIM * C_DIM + 255) / 256, 256, 0, stream>>>(W, Whi, Wlo);

    // 3) fused complex SpMM (edge-parallel, one wave per edge)
    spmm_complex_kernel<<<4096, 256, 0, stream>>>(Xr, Xi, Lr, Li, row, col,
                                                  aggr, aggi, nEdges);

    // 4) WMMA projection + residual  (2N rows, 16-row tiles; 50000 % 16 == 0)
    const int rowTiles = (2 * nNodes + 15) / 16;
    proj_wmma_bf16x3_kernel<<<rowTiles, 256, 0, stream>>>(agg, Whi, Wlo, Xr, Xi,
                                                          out, nNodes);
}
